// Geo_gcn_36833639531234
// MI455X (gfx1250) — matrix operand order repaired and checked by reading, not compiled, not run on hardware
//
#include <hip/hip_runtime.h>

typedef __attribute__((ext_vector_type(16))) _Float16 v16h;
typedef __attribute__((ext_vector_type(8)))  _Float16 v8h;
typedef __attribute__((ext_vector_type(8)))  float    v8f;

#define IN_CH   256
#define OUT_CH  256
#define NJ      25      // valid joints
#define NP      32      // padded joints (2 x 16 tiles)
#define BATCH_N 16384

// LDS leading dimensions (padded: keep 16B alignment, spread banks)
#define XT_LD   264     // xtile   [NP][XT_LD]  f16 row-major (joint, channel)
#define XTT_LD  40      // xtileT  [IN_CH][XTT_LD] f16 (channel, joint)
#define SS_LD   33      // sS      [NP][SS_LD]  f32 scores
#define SP_LD   40      // sP      [NP][SP_LD]  f16 softmax probs
#define AA_LD   264     // aA      [NP][AA_LD]  f16 attention output

// ---- WMMA fragment loaders (CDNA5 ISA 7.12.2 layouts) ----------------------
// A-matrix 16x32 f16: lane m (0..15), half h = lane/16 holds
//   VGPR0-3: K = kBase + h*8 + {0..7};  VGPR4-7: K = kBase + 16 + h*8 + {0..7}
__device__ __forceinline__ v16h load_frag_a(const _Float16* src, int ld,
                                            int rowBase, int kBase) {
  const int lane = threadIdx.x & 31;
  const int m = lane & 15, h = lane >> 4;
  const _Float16* p = src + (rowBase + m) * ld + kBase + h * 8;
  v8h lo = *(const v8h*)(p);        // K: h*8 .. h*8+7
  v8h hi = *(const v8h*)(p + 16);   // K: 16+h*8 .. 16+h*8+7
  v16h r;
#pragma unroll
  for (int i = 0; i < 8; ++i) { r[i] = lo[i]; r[i + 8] = hi[i]; }
  return r;
}

// B-matrix 32x16 f16 (loaded from B^T rows): lane n holds column colBase+n%16,
// half h = lane/16 holds K = kBase + h*8 + {0..7} and kBase + 16 + h*8 + {0..7}
__device__ __forceinline__ v16h load_frag_b(const _Float16* srcT, int ld,
                                            int colBase, int kBase) {
  const int lane = threadIdx.x & 31;
  const int m = lane & 15, h = lane >> 4;
  const _Float16* p = srcT + (colBase + m) * ld + kBase + h * 8;
  v8h lo = *(const v8h*)(p);
  v8h hi = *(const v8h*)(p + 16);
  v16h r;
#pragma unroll
  for (int i = 0; i < 8; ++i) { r[i] = lo[i]; r[i + 8] = hi[i]; }
  return r;
}

// Masked nontemporal store of one 16x16 f32 D-tile (out is write-once).
__device__ __forceinline__ void store_tile(const v8f& c, float* __restrict__ out,
                                           size_t b, int rowBase, int colBase,
                                           int h, int nc) {
#pragma unroll
  for (int r = 0; r < 8; ++r) {
    const int n = rowBase + h * 8 + r;
    if (n < NJ)
      __builtin_nontemporal_store(c[r], &out[(b * NJ + n) * OUT_CH + colBase + nc]);
  }
}

// ---- W transpose + fp32->f16 convert: wt[c][k] = (f16) w[k][c] -------------
__global__ void wt_transpose_kernel(const float* __restrict__ w,
                                    _Float16* __restrict__ wt) {
  const int c = blockIdx.x;    // out channel 0..255
  const int k = threadIdx.x;   // in channel 0..255
  wt[c * IN_CH + k] = (_Float16)w[k * OUT_CH + c];
}

// ---- main fused kernel: one workgroup (8 wave32s) per batch element --------
__global__ __launch_bounds__(256)
void attn_gcn_kernel(const float* __restrict__ x,       // (NJ, BATCH, IN_CH) f32
                     const _Float16* __restrict__ wt,   // (OUT_CH, IN_CH) f16 = W^T
                     float* __restrict__ out) {         // (BATCH, NJ, OUT_CH) f32
  __shared__ _Float16 xtile [NP * XT_LD];
  __shared__ _Float16 xtileT[IN_CH * XTT_LD];
  __shared__ float    sS    [NP * SS_LD];
  __shared__ _Float16 sP    [NP * SP_LD];
  __shared__ _Float16 aA    [NP * AA_LD];

  const int b    = blockIdx.x;
  const int tid  = threadIdx.x;
  const int wave = tid >> 5;
  const int lane = tid & 31;
  const int nc   = lane & 15;
  const int h    = lane >> 4;

  // Warm the shared (L2-resident) weight for this workgroup.
  __builtin_prefetch(wt + tid * ((OUT_CH * IN_CH) / 256), 0, 3);

  // ---- Phase 0: stage x[:, b, :] -> LDS f16 (row-major + transposed) ----
  // x is streamed exactly once -> nontemporal loads keep L2 free for W.
  for (int n = 0; n < NP; ++n) {
    float v = 0.0f;
    if (n < NJ)
      v = __builtin_nontemporal_load(&x[((size_t)n * BATCH_N + b) * IN_CH + tid]);
    _Float16 hv = (_Float16)v;
    xtile [n * XT_LD + tid]   = hv;
    xtileT[tid * XTT_LD + n]  = hv;
  }
  __syncthreads();

  // ---- Phase 1: S = xb * xb^T (32x32 padded), waves 0..3, K=256 ----
  if (wave < 4) {
    const int ti = wave >> 1, tj = wave & 1;
    v8f acc = {};
#pragma unroll
    for (int kb = 0; kb < IN_CH / 32; ++kb) {
      v16h a  = load_frag_a(xtile, XT_LD, ti * 16, kb * 32);
      v16h bt = load_frag_b(xtile, XT_LD, tj * 16, kb * 32); // B^T rows = xb rows
      acc = __builtin_amdgcn_wmma_f32_16x16x32_f16(false, a, false, bt,
                                                   (short)0, acc, false, false);
    }
#pragma unroll
    for (int r = 0; r < 8; ++r)
      sS[(ti * 16 + h * 8 + r) * SS_LD + tj * 16 + nc] = acc[r];
  }
  __syncthreads();

  // ---- Phase 2: row softmax over the 25 valid joints ----
  if (tid < NP) {
    if (tid < NJ) {
      float mx = -3.0e38f;
      for (int m = 0; m < NJ; ++m) mx = fmaxf(mx, sS[tid * SS_LD + m]);
      float e[NJ], sum = 0.0f;
      for (int m = 0; m < NJ; ++m) { e[m] = __expf(sS[tid * SS_LD + m] - mx); sum += e[m]; }
      float inv = 1.0f / sum;
      for (int m = 0;  m < NJ; ++m) sP[tid * SP_LD + m] = (_Float16)(e[m] * inv);
      for (int m = NJ; m < NP; ++m) sP[tid * SP_LD + m] = (_Float16)0.0f;
    } else {
      for (int m = 0; m < NP; ++m) sP[tid * SP_LD + m] = (_Float16)0.0f;
    }
  }
  __syncthreads();

  // ---- Phase 3: A = P(32x32) @ xb(32x256); 32 tiles, 1 WMMA each ----
  {
    const int ti = wave & 1;
    const int tjBase = (wave >> 1) * 4;
    v16h a = load_frag_a(sP, SP_LD, ti * 16, 0);
#pragma unroll
    for (int j = 0; j < 4; ++j) {
      const int tj = tjBase + j;
      v16h bt = load_frag_b(xtileT, XTT_LD, tj * 16, 0); // B^T rows = xb^T rows
      v8f acc = {};
      acc = __builtin_amdgcn_wmma_f32_16x16x32_f16(false, a, false, bt,
                                                   (short)0, acc, false, false);
#pragma unroll
      for (int r = 0; r < 8; ++r)
        aA[(ti * 16 + h * 8 + r) * AA_LD + tj * 16 + nc] = (_Float16)acc[r];
    }
  }
  __syncthreads();

  // ---- Phase 4: out = A(32x256) @ W(256x256) ----
  // Each wave owns a 2x2 block of 16x16 output tiles: both M-halves and
  // columns {2*wave, 2*wave+1}. Every W-fragment (tj,kb) is fetched from L2
  // exactly once per workgroup (128 KB total, no duplication across waves).
  {
    const int tj0 = wave * 2;
    v8f acc00 = {}, acc01 = {}, acc10 = {}, acc11 = {};
#pragma unroll
    for (int kb = 0; kb < 8; ++kb) {
      v16h a0 = load_frag_a(aA, AA_LD, 0,  kb * 32);
      v16h a1 = load_frag_a(aA, AA_LD, 16, kb * 32);
      v16h b0 = load_frag_b(wt, IN_CH, tj0 * 16,       kb * 32);
      v16h b1 = load_frag_b(wt, IN_CH, (tj0 + 1) * 16, kb * 32);
      acc00 = __builtin_amdgcn_wmma_f32_16x16x32_f16(false, a0, false, b0,
                                                     (short)0, acc00, false, false);
      acc01 = __builtin_amdgcn_wmma_f32_16x16x32_f16(false, a0, false, b1,
                                                     (short)0, acc01, false, false);
      acc10 = __builtin_amdgcn_wmma_f32_16x16x32_f16(false, a1, false, b0,
                                                     (short)0, acc10, false, false);
      acc11 = __builtin_amdgcn_wmma_f32_16x16x32_f16(false, a1, false, b1,
                                                     (short)0, acc11, false, false);
    }
    store_tile(acc00, out, (size_t)b, 0,  tj0 * 16,       h, nc);
    store_tile(acc01, out, (size_t)b, 0,  (tj0 + 1) * 16, h, nc);
    store_tile(acc10, out, (size_t)b, 16, tj0 * 16,       h, nc);
    store_tile(acc11, out, (size_t)b, 16, (tj0 + 1) * 16, h, nc);
  }
}

extern "C" void kernel_launch(void* const* d_in, const int* in_sizes, int n_in,
                              void* d_out, int out_size, void* d_ws, size_t ws_size,
                              hipStream_t stream) {
  (void)in_sizes; (void)n_in; (void)out_size; (void)ws_size;
  const float* x = (const float*)d_in[0];
  const float* w = (const float*)d_in[1];
  float* out     = (float*)d_out;
  _Float16* wt   = (_Float16*)d_ws;   // 256*256*2 = 128 KB scratch

  hipLaunchKernelGGL(wt_transpose_kernel, dim3(OUT_CH), dim3(IN_CH), 0, stream, w, wt);
  hipLaunchKernelGGL(attn_gcn_kernel, dim3(BATCH_N), dim3(256), 0, stream, x, wt, out);
}
